// GraphVertResOutModel_23914377904735
// MI455X (gfx1250) — matrix operand order repaired and checked by reading, not compiled, not run on hardware
//
#include <hip/hip_runtime.h>
#include <hip/hip_bf16.h>

typedef __bf16 bf16;
typedef __attribute__((ext_vector_type(16))) __bf16 v16bf;
typedef __attribute__((ext_vector_type(8)))  __bf16 v8bf;
typedef __attribute__((ext_vector_type(8)))  float  v8f;

// ---------------------------------------------------------------------------
// CDNA5 helpers
// ---------------------------------------------------------------------------
__device__ __forceinline__ v16bf ldfrag(const bf16* p0, const bf16* p1) {
  v8bf lo = *(const v8bf*)p0;
  v8bf hi = *(const v8bf*)p1;
  return __builtin_shufflevector(lo, hi, 0,1,2,3,4,5,6,7,8,9,10,11,12,13,14,15);
}

__device__ __forceinline__ v8f wmma_bf16(v16bf a, v16bf b, v8f c) {
  return __builtin_amdgcn_wmma_f32_16x16x32_bf16(false, a, false, b, (short)0, c,
                                                 false, false);
}

// Async global -> LDS copy, 16 bytes per lane (ASYNCcnt-tracked).
// Flat LDS pointers carry the wave-relative LDS byte address in their low 32
// bits (ISA 10.2).
__device__ __forceinline__ void async_stage16(const bf16* lds_dst, const bf16* g_src) {
  unsigned d = (unsigned)(unsigned long long)lds_dst;
  asm volatile("global_load_async_to_lds_b128 %0, %1, off"
               :: "v"(d), "v"(g_src) : "memory");
}
__device__ __forceinline__ void wait_async() {
  asm volatile("s_wait_asynccnt 0x0" ::: "memory");
}

// ---------------------------------------------------------------------------
// f32 -> bf16 conversion (grid-stride)
// ---------------------------------------------------------------------------
__global__ void cvt_kernel(const float* __restrict__ s, bf16* __restrict__ d, int n) {
  int i = blockIdx.x * blockDim.x + threadIdx.x;
  int stride = gridDim.x * blockDim.x;
  for (; i < n; i += stride) d[i] = (bf16)s[i];
}

// ---------------------------------------------------------------------------
// Out = act( A @ W^T (+bias) (+Res) ),  N = K = 128.
// Block: 256 thr = 8 waves, block tile 256(M) x 128(N); wave tile 32 x 128
// (two M sub-tiles share every B fragment -> ~1.25 ds_load per WMMA).
// ---------------------------------------------------------------------------
template<bool RELU, bool BIAS, bool RES>
__global__ __launch_bounds__(256)
void gemm128_kernel(const bf16* __restrict__ A, const bf16* __restrict__ W,
                    const float* __restrict__ bias, const bf16* Res, bf16* Out)
{
  extern __shared__ __align__(16) char smraw[];
  bf16* Xs = (bf16*)smraw;            // [256][136]  (68-dword stride, conflict-free)
  bf16* Ws = Xs + 256 * 136;          // [128][136]
  const int m0   = blockIdx.x << 8;
  const int t    = threadIdx.x;
  const int wave = t >> 5, lane = t & 31;
  const int half = lane >> 4, l16 = lane & 15;

  // async staging: X tile 256x128 (4096 16B chunks), W 128x128 (2048 chunks)
#pragma unroll
  for (int i = 0; i < 16; ++i) {
    int c  = t + (i << 8);
    int r  = c >> 4;
    int cc = (c & 15) << 3;
    async_stage16(&Xs[r * 136 + cc], &A[(size_t)(m0 + r) * 128 + cc]);
  }
#pragma unroll
  for (int i = 0; i < 8; ++i) {
    int c  = t + (i << 8);
    int r  = c >> 4;
    int cc = (c & 15) << 3;
    async_stage16(&Ws[r * 136 + cc], &W[(size_t)r * 128 + cc]);
  }
  wait_async();
  __syncthreads();

  v8f acc0[8] = {}, acc1[8] = {};
  const int r0 = wave * 32 + l16;
  const int r1 = r0 + 16;
#pragma unroll
  for (int k0 = 0; k0 < 128; k0 += 32) {
    const bf16* a0p = &Xs[r0 * 136 + k0 + half * 8];
    const bf16* a1p = &Xs[r1 * 136 + k0 + half * 8];
    v16bf a0 = ldfrag(a0p, a0p + 16);
    v16bf a1 = ldfrag(a1p, a1p + 16);
#pragma unroll
    for (int nt = 0; nt < 8; ++nt) {
      const bf16* bp = &Ws[(nt * 16 + l16) * 136 + k0 + half * 8];
      v16bf b = ldfrag(bp, bp + 16);
      acc0[nt] = wmma_bf16(a0, b, acc0[nt]);
      acc1[nt] = wmma_bf16(a1, b, acc1[nt]);
    }
  }

#pragma unroll
  for (int nt = 0; nt < 8; ++nt) {
    const int ncol = nt * 16 + l16;
    float bv = 0.f;
    if (BIAS) bv = bias[ncol];
#pragma unroll
    for (int j = 0; j < 8; ++j) {
      {
        const size_t midx = (size_t)(m0 + wave * 32 + half * 8 + j) * 128 + ncol;
        float v = acc0[nt][j] + bv;
        if (RES) v += (float)Res[midx];
        if (RELU) v = v > 0.f ? v : 0.f;
        Out[midx] = (bf16)v;
      }
      {
        const size_t midx = (size_t)(m0 + wave * 32 + 16 + half * 8 + j) * 128 + ncol;
        float v = acc1[nt][j] + bv;
        if (RES) v += (float)Res[midx];
        if (RELU) v = v > 0.f ? v : 0.f;
        Out[midx] = (bf16)v;
      }
    }
  }
}

// ---------------------------------------------------------------------------
// x_next = relu( G0[b] (256x256) @ y[b] (256x128) ).  One block per batch.
// Block tile 256(M) x 128(N), K = 256; wave tile 32 x 128.
// y staged TRANSPOSED in LDS so B fragments are contiguous reads.
// ---------------------------------------------------------------------------
__global__ __launch_bounds__(256)
void graph_agg_kernel(const bf16* __restrict__ G, const bf16* __restrict__ Y,
                      bf16* __restrict__ Xn)
{
  extern __shared__ __align__(16) char smraw[];
  bf16* Gs = (bf16*)smraw;            // [256][264]
  bf16* Ys = Gs + 256 * 264;          // yT: [128 p][264 n]
  const int b = blockIdx.x;
  const bf16* Gb = G + (size_t)b * 65536;
  const bf16* Yb = Y + (size_t)b * 32768;
  const int t = threadIdx.x;
  const int wave = t >> 5, lane = t & 31;
  const int half = lane >> 4, l16 = lane & 15;

  // stage G (256x256 = 8192 chunks) via async-to-LDS
#pragma unroll
  for (int i = 0; i < 32; ++i) {
    int c  = t + (i << 8);
    int r  = c >> 5;
    int cc = (c & 31) << 3;
    async_stage16(&Gs[r * 264 + cc], &Gb[(size_t)r * 256 + cc]);
  }
  // stage y transposed: Ys[p][n] = y[n][p]
#pragma unroll 8
  for (int i = 0; i < 128; ++i) {
    int idx = t + (i << 8);
    int n = idx >> 7, p = idx & 127;
    Ys[p * 264 + n] = Yb[idx];
  }
  wait_async();
  __syncthreads();

  v8f acc0[8] = {}, acc1[8] = {};
  const int r0 = wave * 32 + l16;
  const int r1 = r0 + 16;
#pragma unroll
  for (int k0 = 0; k0 < 256; k0 += 32) {
    const bf16* a0p = &Gs[r0 * 264 + k0 + half * 8];
    const bf16* a1p = &Gs[r1 * 264 + k0 + half * 8];
    v16bf a0 = ldfrag(a0p, a0p + 16);
    v16bf a1 = ldfrag(a1p, a1p + 16);
#pragma unroll
    for (int nt = 0; nt < 8; ++nt) {
      const bf16* bp = &Ys[(nt * 16 + l16) * 264 + k0 + half * 8];
      v16bf bb = ldfrag(bp, bp + 16);
      acc0[nt] = wmma_bf16(a0, bb, acc0[nt]);
      acc1[nt] = wmma_bf16(a1, bb, acc1[nt]);
    }
  }

  bf16* Ob = Xn + (size_t)b * 32768;
#pragma unroll
  for (int nt = 0; nt < 8; ++nt) {
    const int ncol = nt * 16 + l16;
#pragma unroll
    for (int j = 0; j < 8; ++j) {
      float v0 = acc0[nt][j];
      v0 = v0 > 0.f ? v0 : 0.f;
      Ob[(size_t)(wave * 32 + half * 8 + j) * 128 + ncol] = (bf16)v0;
      float v1 = acc1[nt][j];
      v1 = v1 > 0.f ? v1 : 0.f;
      Ob[(size_t)(wave * 32 + 16 + half * 8 + j) * 128 + ncol] = (bf16)v1;
    }
  }
}

// ---------------------------------------------------------------------------
// Fused head: out[m] = relu(H[m] @ f1_W^T + f1_b) . f2_w + f2_b
// 1024 N-cols in 8 chunks of 128; per-row dot kept in registers; reduced
// with __shfl_xor butterflies inside each 16-lane half (wave32).
// ---------------------------------------------------------------------------
__global__ __launch_bounds__(256)
void head_kernel(const bf16* __restrict__ H,  const bf16* __restrict__ W1,
                 const float* __restrict__ b1, const float* __restrict__ w2,
                 const float* __restrict__ b2, float* __restrict__ Out)
{
  extern __shared__ __align__(16) char smraw[];
  bf16* Xs = (bf16*)smraw;            // [128][136]
  bf16* Ws = Xs + 128 * 136;          // [128][136]
  const int m0 = blockIdx.x << 7;
  const int t = threadIdx.x;
  const int wave = t >> 5, lane = t & 31;
  const int half = lane >> 4, l16 = lane & 15;

#pragma unroll
  for (int i = 0; i < 8; ++i) {
    int c = t + (i << 8);
    int r = c >> 4, cc = (c & 15) << 3;
    async_stage16(&Xs[r * 136 + cc], &H[(size_t)(m0 + r) * 128 + cc]);
  }

  float rowsum[8] = {0.f, 0.f, 0.f, 0.f, 0.f, 0.f, 0.f, 0.f};
  const int rowA = wave * 16 + l16;

  for (int ch = 0; ch < 8; ++ch) {
    __syncthreads();                  // previous chunk's reads done
#pragma unroll
    for (int i = 0; i < 8; ++i) {
      int c = t + (i << 8);
      int r = c >> 4, cc = (c & 15) << 3;
      async_stage16(&Ws[r * 136 + cc], &W1[(size_t)(ch * 128 + r) * 128 + cc]);
    }
    wait_async();                     // also covers Xs staging on ch==0
    __syncthreads();

    v8f acc[8] = {};
#pragma unroll
    for (int k0 = 0; k0 < 128; k0 += 32) {
      const bf16* ap = &Xs[rowA * 136 + k0 + half * 8];
      v16bf a = ldfrag(ap, ap + 16);
#pragma unroll
      for (int nt = 0; nt < 8; ++nt) {
        const bf16* bp = &Ws[(nt * 16 + l16) * 136 + k0 + half * 8];
        v16bf b = ldfrag(bp, bp + 16);
        acc[nt] = wmma_bf16(a, b, acc[nt]);
      }
    }
#pragma unroll
    for (int nt = 0; nt < 8; ++nt) {
      const int ncol = ch * 128 + nt * 16 + l16;
      const float bv = b1[ncol];
      const float wv = w2[ncol];
#pragma unroll
      for (int j = 0; j < 8; ++j) {
        float v = acc[nt][j] + bv;
        v = v > 0.f ? v : 0.f;
        rowsum[j] += v * wv;
      }
    }
  }

#pragma unroll
  for (int off = 1; off < 16; off <<= 1) {
#pragma unroll
    for (int j = 0; j < 8; ++j)
      rowsum[j] += __shfl_xor(rowsum[j], off, 32);
  }

  if (l16 == 0) {
    const float bb = b2[0];
#pragma unroll
    for (int j = 0; j < 8; ++j)
      Out[m0 + wave * 16 + half * 8 + j] = rowsum[j] + bb;
  }
}

// ---------------------------------------------------------------------------
// Host-side orchestration
// ---------------------------------------------------------------------------
extern "C" void kernel_launch(void* const* d_in, const int* in_sizes, int n_in,
                              void* d_out, int out_size, void* d_ws, size_t ws_size,
                              hipStream_t stream)
{
  (void)in_sizes; (void)n_in; (void)out_size; (void)ws_size;
  const float* G     = (const float*)d_in[0];
  const float* xG    = (const float*)d_in[1];
  const float* gmlW  = (const float*)d_in[2];
  const float* gmlB  = (const float*)d_in[3];
  const float* lin0W = (const float*)d_in[4];
  const float* lin0B = (const float*)d_in[5];
  const float* rinW  = (const float*)d_in[6];
  const float* rinB  = (const float*)d_in[7];
  const float* rb1W  = (const float*)d_in[8];
  const float* rb1B  = (const float*)d_in[9];
  const float* rb2W  = (const float*)d_in[10];
  const float* routW = (const float*)d_in[11];
  const float* routB = (const float*)d_in[12];
  const float* f1W   = (const float*)d_in[13];
  const float* f1B   = (const float*)d_in[14];
  const float* f2W   = (const float*)d_in[15];
  const float* f2B   = (const float*)d_in[16];
  float* out = (float*)d_out;

  char* ws = (char*)d_ws;
  size_t off = 0;
  auto alloc = [&](size_t bytes) -> bf16* {
    bf16* p = (bf16*)(ws + off);
    off = (off + bytes + 255) & ~(size_t)255;
    return p;
  };
  bf16* Gb   = alloc((size_t)512 * 256 * 256 * 2);  // 67 MB (L2-resident)
  bf16* Xb   = alloc((size_t)512 * 256 * 128 * 2);  // ping
  bf16* Yb   = alloc((size_t)512 * 256 * 128 * 2);  // pong
  bf16* Wgml = alloc(4 * 128 * 128 * 2);
  bf16* Wl0  = alloc(128 * 128 * 2);
  bf16* Wri  = alloc(128 * 128 * 2);
  bf16* Wb1  = alloc(3 * 128 * 128 * 2);
  bf16* Wb2  = alloc(3 * 128 * 128 * 2);
  bf16* Wro  = alloc(128 * 128 * 2);
  bf16* Wf1  = alloc(1024 * 128 * 2);

  auto cvt = [&](const float* s, bf16* d, int n) {
    int blocks = (n + 1023) >> 10;
    if (blocks > 16384) blocks = 16384;
    if (blocks < 1) blocks = 1;
    cvt_kernel<<<blocks, 256, 0, stream>>>(s, d, n);
  };
  cvt(G,     Gb,   512 * 256 * 256);
  cvt(xG,    Xb,   512 * 256 * 128);
  cvt(gmlW,  Wgml, 4 * 128 * 128);
  cvt(lin0W, Wl0,  128 * 128);
  cvt(rinW,  Wri,  128 * 128);
  cvt(rb1W,  Wb1,  3 * 128 * 128);
  cvt(rb2W,  Wb2,  3 * 128 * 128);
  cvt(routW, Wro,  128 * 128);
  cvt(f1W,   Wf1,  1024 * 128);

  const int gblocks = (512 * 256) / 256;                        // 512
  const size_t gemmSh = (256 * 136 + 128 * 136) * sizeof(bf16); // 104448 B
  const size_t aggSh  = (256 * 264 + 128 * 264) * sizeof(bf16); // 202752 B
  const size_t headSh = (128 * 136 + 128 * 136) * sizeof(bf16); // 69632 B

  // ---- graph message-passing layers ----
  for (int l = 0; l < 4; ++l) {
    gemm128_kernel<false, true, false><<<gblocks, 256, gemmSh, stream>>>(
        Xb, Wgml + (size_t)l * 128 * 128, gmlB + l * 128, nullptr, Yb);
    graph_agg_kernel<<<512, 256, aggSh, stream>>>(Gb, Yb, Xb);
  }

  // ---- trunk ----
  gemm128_kernel<false, true, false><<<gblocks, 256, gemmSh, stream>>>(
      Xb, Wl0, lin0B, nullptr, Yb);
  gemm128_kernel<false, true, false><<<gblocks, 256, gemmSh, stream>>>(
      Yb, Wri, rinB, nullptr, Xb);
  for (int i = 0; i < 3; ++i) {
    gemm128_kernel<true, true, false><<<gblocks, 256, gemmSh, stream>>>(
        Xb, Wb1 + (size_t)i * 128 * 128, rb1B + i * 128, nullptr, Yb);
    gemm128_kernel<false, false, true><<<gblocks, 256, gemmSh, stream>>>(
        Yb, Wb2 + (size_t)i * 128 * 128, nullptr, Xb, Xb);
  }
  gemm128_kernel<false, true, false><<<gblocks, 256, gemmSh, stream>>>(
      Xb, Wro, routB, nullptr, Yb);

  // ---- fused head ----
  head_kernel<<<(512 * 256) / 128, 256, headSh, stream>>>(Yb, Wf1, f1B, f2W, f2B, out);
}
